// AttentionMaskGenerator_80582176408039
// MI455X (gfx1250) — compile-verified, matrix operand order checked by reading, not run on hardware
//
#include <hip/hip_runtime.h>
#include <hip/hip_bf16.h>

#define D_  1024
#define H_  1024
#define H2_ 512
#define M_  15
#define B_  8192
#define ROWS_ 64

typedef __bf16 bf16;
typedef __attribute__((ext_vector_type(8)))  bf16  v8bf;
typedef __attribute__((ext_vector_type(16))) bf16  v16bf;
typedef __attribute__((ext_vector_type(8)))  float v8f;

// ---- WMMA helpers -----------------------------------------------------------
// 16-bit A/B fragment (16x32 MxK; for B, NxK of Wt): per CDNA5 ISA 7.12.2,
// lane L covers row (L&15); K order per lane: [kh*8.., 16+kh*8..], kh = L>>4.
__device__ __forceinline__ v16bf load_frag(const bf16* __restrict__ base, int ld, int lane) {
  const int row = lane & 15;
  const int kh  = (lane >> 4) & 1;
  const bf16* p = base + (size_t)row * ld + kh * 8;
  v8bf lo = *(const v8bf*)p;
  v8bf hi = *(const v8bf*)(p + 16);
  return __builtin_shufflevector(lo, hi, 0,1,2,3,4,5,6,7,8,9,10,11,12,13,14,15);
}

__device__ __forceinline__ v8f wmma_bf16(v16bf a, v16bf b, v8f c) {
  return __builtin_amdgcn_wmma_f32_16x16x32_bf16(false, a, false, b, (short)0, c, false, false);
}

__device__ __forceinline__ float gelu_exact(float x) {
  return 0.5f * x * (1.0f + erff(x * 0.70710678118654752440f));
}

// ---- prep: fp32 -> bf16 convert --------------------------------------------
__global__ void cvt_f32_bf16(const float* __restrict__ in, bf16* __restrict__ out, int n) {
  int i = blockIdx.x * blockDim.x + threadIdx.x;
  if (i < n) out[i] = (bf16)in[i];
}

// ---- prep: transpose [R,C] fp32 -> [C,R] bf16, batched over blockIdx.z -----
__global__ __launch_bounds__(256)
void transpose_cvt(const float* __restrict__ in, bf16* __restrict__ out, int R, int C) {
  __shared__ float tile[32][33];
  const float* src = in  + (size_t)blockIdx.z * R * C;
  bf16*        dst = out + (size_t)blockIdx.z * R * C;
  const int c0 = blockIdx.x * 32, r0 = blockIdx.y * 32;
  const int tx = threadIdx.x, ty = threadIdx.y;
  for (int i = ty; i < 32; i += 8)
    tile[i][tx] = src[(size_t)(r0 + i) * C + c0 + tx];
  __syncthreads();
  for (int i = ty; i < 32; i += 8)
    dst[(size_t)(c0 + i) * R + r0 + tx] = (bf16)tile[tx][i];
}

// ---- WMMA GEMM (64-row x 256-col blocks): Out = A.Wt^T + bias --------------
// 512 threads / 16 waves; each wave: 4 row-blocks x 1 n-tile, B-frag reuse x4.
__global__ __launch_bounds__(512)
void gemm_bias_bf16(const bf16* __restrict__ A, const bf16* __restrict__ Wt,
                    const float* __restrict__ bias, bf16* __restrict__ Out,
                    int K, int Nout) {
  __shared__ bf16 sA[64 * 1024];   // 128KB
  const int tid = threadIdx.x, wave = tid >> 5, lane = tid & 31;
  const int rowBase = blockIdx.x * 64;
  {
    const uint4* src = (const uint4*)(A + (size_t)rowBase * K);
    uint4* dst = (uint4*)sA;
    for (int i = tid; i < 64 * K / 8; i += 512) dst[i] = src[i];
  }
  __syncthreads();
  const int n = blockIdx.y * 256 + wave * 16;
  v8f acc[4] = {};
  for (int kb = 0; kb < K; kb += 32) {
    v16bf a[4];
    #pragma unroll
    for (int rb = 0; rb < 4; ++rb) a[rb] = load_frag(sA + rb * 16 * K + kb, K, lane);
    v16bf wf = load_frag(Wt + (size_t)n * K + kb, K, lane);
    #pragma unroll
    for (int rb = 0; rb < 4; ++rb) acc[rb] = wmma_bf16(a[rb], wf, acc[rb]);
  }
  const int col  = n + (lane & 15);
  const float bc = bias[col];
  const int roff = (lane >> 4) * 8;
  #pragma unroll
  for (int rb = 0; rb < 4; ++rb)
    #pragma unroll
    for (int r = 0; r < 8; ++r)
      Out[(size_t)(rowBase + rb * 16 + roff + r) * Nout + col] = (bf16)(acc[rb][r] + bc);
}

// ---- fused 15-mask MLP: GEMM1+LN+GELU -> GEMM2+GELU -> GEMM3+sigmoid -------
// 64-row tiles, 512 threads / 16 waves: every weight fragment loaded once per
// wave is reused for 4 row-blocks in registers -> weight streaming from L2 is
// (B/64) * (W1+W2+W3) per mask (~7.7 GB total), 4x less than 16-row tiling.
__global__ __launch_bounds__(512)
void mask_mlp_fused(const bf16* __restrict__ Abf,
                    const bf16* __restrict__ W1t, const bf16* __restrict__ W2t,
                    const bf16* __restrict__ W3t,
                    const float* __restrict__ b1, const float* __restrict__ lng,
                    const float* __restrict__ lnb, const float* __restrict__ b2,
                    const float* __restrict__ b3, float* __restrict__ Out) {
  __shared__ __align__(16) unsigned char smem[256 * 1024];
  bf16* sA   = (bf16*)smem;                  // 128KB: A tile during GEMM1
  bf16* sH2b = (bf16*)smem;                  //  64KB: h2 (aliases dead sA)
  bf16* sH1b = (bf16*)(smem + 128 * 1024);   // 128KB: h1

  const int tid = threadIdx.x, wave = tid >> 5, lane = tid & 31;
  const int m = blockIdx.y;
  const int rowBase = blockIdx.x * ROWS_;
  const int roff = (lane >> 4) * 8;

  { // stage 64x1024 bf16 A tile (128KB)
    const uint4* src = (const uint4*)(Abf + (size_t)rowBase * D_);
    uint4* dst = (uint4*)sA;
    #pragma unroll
    for (int i = 0; i < 16; ++i) dst[tid + 512 * i] = src[tid + 512 * i];
  }
  __syncthreads();

  // GEMM1: 64 N-tiles of H; per wave 4 n-tiles x 4 row-blocks.
  {
    const bf16* W1m = W1t + (size_t)m * H_ * D_;
    v8f acc[4][4] = {};
    for (int kb = 0; kb < D_; kb += 32) {
      v16bf a[4];
      #pragma unroll
      for (int rb = 0; rb < 4; ++rb) a[rb] = load_frag(sA + rb * 16 * D_ + kb, D_, lane);
      #pragma unroll
      for (int j = 0; j < 4; ++j) {
        const int n = (wave * 4 + j) * 16;
        v16bf wf = load_frag(W1m + (size_t)n * D_ + kb, D_, lane);
        #pragma unroll
        for (int rb = 0; rb < 4; ++rb) acc[rb][j] = wmma_bf16(a[rb], wf, acc[rb][j]);
      }
    }
    #pragma unroll
    for (int rb = 0; rb < 4; ++rb)
      #pragma unroll
      for (int j = 0; j < 4; ++j) {
        const int col = (wave * 4 + j) * 16 + (lane & 15);
        const float bias = b1[(size_t)m * H_ + col];
        #pragma unroll
        for (int r = 0; r < 8; ++r)
          sH1b[(rb * 16 + roff + r) * H_ + col] = (bf16)(acc[rb][j][r] + bias);
      }
  }
  __syncthreads();

  // LayerNorm over H + exact GELU, in place; 4 rows per wave (16 waves x 4 = 64).
  #pragma unroll
  for (int rr = 0; rr < 4; ++rr) {
    const int row = wave * 4 + rr;
    float s = 0.f, s2 = 0.f;
    for (int c = lane; c < H_; c += 32) {
      float v = (float)sH1b[row * H_ + c];
      s += v; s2 += v * v;
    }
    #pragma unroll
    for (int off = 16; off > 0; off >>= 1) {
      s  += __shfl_xor(s,  off, 32);
      s2 += __shfl_xor(s2, off, 32);
    }
    const float mu   = s * (1.f / H_);
    const float rstd = rsqrtf(s2 * (1.f / H_) - mu * mu + 1e-5f);
    for (int c = lane; c < H_; c += 32) {
      float v = ((float)sH1b[row * H_ + c] - mu) * rstd;
      v = v * lng[(size_t)m * H_ + c] + lnb[(size_t)m * H_ + c];
      sH1b[row * H_ + c] = (bf16)gelu_exact(v);
    }
  }
  __syncthreads();

  // GEMM2: 32 N-tiles of H2; per wave 2 n-tiles x 4 row-blocks.
  {
    const bf16* W2m = W2t + (size_t)m * H2_ * H_;
    v8f acc[4][2] = {};
    for (int kb = 0; kb < H_; kb += 32) {
      v16bf a[4];
      #pragma unroll
      for (int rb = 0; rb < 4; ++rb) a[rb] = load_frag(sH1b + rb * 16 * H_ + kb, H_, lane);
      #pragma unroll
      for (int j = 0; j < 2; ++j) {
        const int n = (wave * 2 + j) * 16;
        v16bf wf = load_frag(W2m + (size_t)n * H_ + kb, H_, lane);
        #pragma unroll
        for (int rb = 0; rb < 4; ++rb) acc[rb][j] = wmma_bf16(a[rb], wf, acc[rb][j]);
      }
    }
    #pragma unroll
    for (int rb = 0; rb < 4; ++rb)
      #pragma unroll
      for (int j = 0; j < 2; ++j) {
        const int col = (wave * 2 + j) * 16 + (lane & 15);
        const float bias = b2[(size_t)m * H2_ + col];
        #pragma unroll
        for (int r = 0; r < 8; ++r)
          sH2b[(rb * 16 + roff + r) * H2_ + col] = (bf16)gelu_exact(acc[rb][j][r] + bias);
      }
  }
  __syncthreads();

  // GEMM3: 64 N-tiles of D; per wave 4 n-tiles x 4 row-blocks; sigmoid -> NT store.
  {
    const bf16* W3m = W3t + (size_t)m * D_ * H2_;
    float* OutM = Out + ((size_t)m * B_ + rowBase) * D_;
    v8f acc[4][4] = {};
    for (int kb = 0; kb < H2_; kb += 32) {
      v16bf a[4];
      #pragma unroll
      for (int rb = 0; rb < 4; ++rb) a[rb] = load_frag(sH2b + rb * 16 * H2_ + kb, H2_, lane);
      #pragma unroll
      for (int j = 0; j < 4; ++j) {
        const int n = (wave * 4 + j) * 16;
        v16bf wf = load_frag(W3m + (size_t)n * H2_ + kb, H2_, lane);
        #pragma unroll
        for (int rb = 0; rb < 4; ++rb) acc[rb][j] = wmma_bf16(a[rb], wf, acc[rb][j]);
      }
    }
    #pragma unroll
    for (int rb = 0; rb < 4; ++rb)
      #pragma unroll
      for (int j = 0; j < 4; ++j) {
        const int col = (wave * 4 + j) * 16 + (lane & 15);
        const float bias = b3[(size_t)m * D_ + col];
        #pragma unroll
        for (int r = 0; r < 8; ++r) {
          const float v = acc[rb][j][r] + bias;
          __builtin_nontemporal_store(1.f / (1.f + expf(-v)),
              &OutM[(size_t)(rb * 16 + roff + r) * D_ + col]);
        }
      }
  }
}

// ---- host -------------------------------------------------------------------
extern "C" void kernel_launch(void* const* d_in, const int* in_sizes, int n_in,
                              void* d_out, int out_size, void* d_ws, size_t ws_size,
                              hipStream_t stream) {
  (void)in_sizes; (void)n_in; (void)out_size; (void)ws_size;
  const float* x   = (const float*)d_in[0];
  const float* ipw = (const float*)d_in[1];
  const float* ipb = (const float*)d_in[2];
  const float* opw = (const float*)d_in[3];
  const float* opb = (const float*)d_in[4];
  const float* W1  = (const float*)d_in[5];
  const float* b1  = (const float*)d_in[6];
  const float* lng = (const float*)d_in[7];
  const float* lnb = (const float*)d_in[8];
  const float* W2  = (const float*)d_in[9];
  const float* b2  = (const float*)d_in[10];
  const float* W3  = (const float*)d_in[11];
  const float* b3  = (const float*)d_in[12];
  float* out = (float*)d_out;

  unsigned char* ws = (unsigned char*)d_ws;
  size_t off = 0;
  auto alloc = [&](size_t bytes) -> bf16* {
    bf16* p = (bf16*)(ws + off);
    off += (bytes + 255) & ~(size_t)255;
    return p;
  };
  bf16* x_bf  = alloc((size_t)B_ * D_ * 2);          // 16 MB
  bf16* v_bf  = alloc((size_t)B_ * D_ * 2);          // 16 MB
  bf16* a_bf  = alloc((size_t)B_ * D_ * 2);          // 16 MB
  bf16* Wv_bf = alloc((size_t)D_ * D_ * 2);          //  2 MB
  bf16* Wo_bf = alloc((size_t)D_ * D_ * 2);          //  2 MB
  bf16* W1t   = alloc((size_t)M_ * H_  * D_  * 2);   // 30 MB  [M,H,D]
  bf16* W2t   = alloc((size_t)M_ * H2_ * H_  * 2);   // 15 MB  [M,H2,H]
  bf16* W3t   = alloc((size_t)M_ * D_  * H2_ * 2);   // 15 MB  [M,D,H2]
  // total ~112 MB of d_ws

  // prep: converts + transposes (softmax over len-1 == 1, so only V/out proj matter)
  cvt_f32_bf16<<<(B_ * D_) / 256, 256, 0, stream>>>(x, x_bf, B_ * D_);
  cvt_f32_bf16<<<(D_ * D_) / 256, 256, 0, stream>>>(ipw + (size_t)2 * D_ * D_, Wv_bf, D_ * D_);
  cvt_f32_bf16<<<(D_ * D_) / 256, 256, 0, stream>>>(opw, Wo_bf, D_ * D_);
  transpose_cvt<<<dim3(H_ / 32,  D_ / 32,  M_), dim3(32, 8), 0, stream>>>(W1, W1t, D_,  H_);
  transpose_cvt<<<dim3(H2_ / 32, H_ / 32,  M_), dim3(32, 8), 0, stream>>>(W2, W2t, H_,  H2_);
  transpose_cvt<<<dim3(D_ / 32,  H2_ / 32, M_), dim3(32, 8), 0, stream>>>(W3, W3t, H2_, D_);

  // attention collapses to two GEMMs: v = x.Wv^T + bv ; a = v.Wo^T + bo
  gemm_bias_bf16<<<dim3(B_ / 64, D_ / 256), 512, 0, stream>>>(x_bf, Wv_bf, ipb + 2 * D_, v_bf, D_, D_);
  gemm_bias_bf16<<<dim3(B_ / 64, D_ / 256), 512, 0, stream>>>(v_bf, Wo_bf, opb, a_bf, D_, D_);

  // fused 15-mask MLP stack, h1/h2 never touch HBM
  mask_mlp_fused<<<dim3(B_ / ROWS_, M_), 512, 0, stream>>>(a_bf, W1t, W2t, W3t,
                                                           b1, lng, lnb, b2, b3, out);
}
